// GatedConvSSM3D_19628000543411
// MI455X (gfx1250) — matrix operand order repaired and checked by reading, not compiled, not run on hardware
//
#include <hip/hip_runtime.h>
#include <hip/hip_bf16.h>

// ---------------- problem constants ----------------
#define Bsz   2
#define Tsz   32
#define Hsz   64
#define Wsz   64
#define Csz   64
#define KT    5
#define KS    7
#define NTAPS (KT*KS*KS)          // 245
#define NPTS  (Bsz*Tsz*Hsz*Wsz)  // 262144 rows
#define NC    ((size_t)NPTS*Csz) // 16,777,216 elements per buffer

typedef __attribute__((ext_vector_type(16))) _Float16 v16h;
typedef __attribute__((ext_vector_type(8)))  float    v8f;

// =====================================================================
// Fused gate/hidden projection:  gp = x@gw+gb, hp = x@hw+hb
// WMMA f32_16x16x32_f16, one wave handles 8 row-tiles of 16 rows,
// keeping all B fragments (both matrices) resident in VGPRs.
// =====================================================================
__global__ __launch_bounds__(256) void proj2_kernel(
    const float* __restrict__ x,
    const float* __restrict__ gw, const float* __restrict__ gb,
    const float* __restrict__ hw, const float* __restrict__ hb,
    float* __restrict__ gp, float* __restrict__ hp)
{
    const int lane  = threadIdx.x & 31;
    const int wave  = threadIdx.x >> 5;
    const int gwv   = blockIdx.x * 8 + wave;   // 2048 waves total
    const int tile0 = gwv * 8;                 // 8 row-tiles per wave
    const int ncol  = lane & 15;
    const int hi    = lane >> 4;

    // ---- load B fragments for both weights: [ks][nt] ----
    // B fragment (32x16), contiguous k order as in A: lane col N, K=klo..klo+15
    v16h Bg[2][4], Bh[2][4];
    for (int ks = 0; ks < 2; ++ks) {
        const int klo = hi * 16 + ks * 32;
        for (int nt = 0; nt < 4; ++nt) {
            const int n = ncol + nt * 16;
            v16h bg, bh;
#pragma unroll
            for (int j = 0; j < 16; ++j) {
                bg[j] = (_Float16)gw[(size_t)(klo + j) * Csz + n];
                bh[j] = (_Float16)hw[(size_t)(klo + j) * Csz + n];
            }
            Bg[ks][nt] = bg; Bh[ks][nt] = bh;
        }
    }
    float gbias[4], hbias[4];
#pragma unroll
    for (int nt = 0; nt < 4; ++nt) {
        gbias[nt] = gb[ncol + nt * 16];
        hbias[nt] = hb[ncol + nt * 16];
    }

    for (int r = 0; r < 8; ++r) {
        const int rowtile = tile0 + r;
        const size_t row  = (size_t)rowtile * 16 + ncol;
        const float* xr   = x + row * Csz;

        v8f accg[4], acch[4];
#pragma unroll
        for (int nt = 0; nt < 4; ++nt)
#pragma unroll
            for (int i = 0; i < 8; ++i) { accg[nt][i] = 0.f; acch[nt][i] = 0.f; }

#pragma unroll
        for (int ks = 0; ks < 2; ++ks) {
            // A fragment (16x32), contiguous k order as in B: lane half hi: K = hi*16 + {0..15}
            const int kb = hi * 16 + ks * 32;
            v16h a;
#pragma unroll
            for (int j = 0; j < 16; ++j) {
                const int k = kb + j;
                a[j] = (_Float16)xr[k];
            }
#pragma unroll
            for (int nt = 0; nt < 4; ++nt) {
                accg[nt] = __builtin_amdgcn_wmma_f32_16x16x32_f16(
                    false, a, false, Bg[ks][nt], (short)0, accg[nt], false, false);
                acch[nt] = __builtin_amdgcn_wmma_f32_16x16x32_f16(
                    false, a, false, Bh[ks][nt], (short)0, acch[nt], false, false);
            }
        }

        // D layout: VGPR v -> row m = v + 8*hi, col n = ncol + nt*16
        const size_t mbase = (size_t)rowtile * 16 + hi * 8;
#pragma unroll
        for (int nt = 0; nt < 4; ++nt) {
            const int n = ncol + nt * 16;
#pragma unroll
            for (int v = 0; v < 8; ++v) {
                const size_t m = mbase + v;
                gp[m * Csz + n] = accg[nt][v] + gbias[nt];
                hp[m * Csz + n] = acch[nt][v] + hbias[nt];
            }
        }
    }
}

// =====================================================================
// Depthwise circular 3D convolution, 5x7x7 taps, per (b,c) plane.
// out[t,h,w] = sum_{dt,dh,dw} kern[dt,dh,dw] * in[(t-dt+2)%T,(h-dh+3)%H,(w-dw+3)%W]
// LDS tile: (4+4) x (16+6) x (64+6) floats = 49.3 KB, filled with
// CDNA5 GLOBAL_LOAD_ASYNC_TO_LDS_B32 (no VGPR round-trip, ASYNCcnt).
// =====================================================================
#define TT 4
#define HT 16
#define TP (TT + 4)
#define HP (HT + 6)
#define WP (Wsz + 6)

__global__ __launch_bounds__(256) void dwconv_kernel(
    const float* __restrict__ src,   // (B,T,H,W,C)
    const float* __restrict__ kern,  // (C,KT,KS,KS)
    float* __restrict__ dst)         // (B,T,H,W,C)
{
    __shared__ float tileS[TP * HP * WP];
    __shared__ float kS[NTAPS];

    const int tid = threadIdx.x;
    const int bc  = blockIdx.y;          // B*C = 128
    const int b   = bc >> 6;
    const int c   = bc & 63;
    const int tl0 = blockIdx.x >> 2;     // 8 t-tiles
    const int hl0 = blockIdx.x & 3;      // 4 h-tiles
    const int t0  = tl0 * TT;
    const int h0  = hl0 * HT;

    for (int i = tid; i < NTAPS; i += 256)
        kS[i] = kern[(size_t)c * NTAPS + i];

    const int total = TP * HP * WP;      // 12320
    for (int i = tid; i < total; i += 256) {
        const int tl  = i / (HP * WP);
        const int rem = i % (HP * WP);
        const int hl  = rem / WP;
        const int wl  = rem % WP;
        const int t = (t0 + tl - 2) & (Tsz - 1);
        const int h = (h0 + hl - 3) & (Hsz - 1);
        const int w = (wl - 3) & (Wsz - 1);
        const float* gsrc =
            &src[((((size_t)b * Tsz + t) * Hsz + h) * Wsz + w) * Csz + c];
        // low 32 bits of the flat shared-aperture address == LDS byte offset
        const unsigned int ldsoff = (unsigned int)(uintptr_t)(&tileS[i]);
        asm volatile("global_load_async_to_lds_b32 %0, %1, off"
                     :: "v"(ldsoff), "v"(gsrc) : "memory");
    }
    asm volatile("s_wait_asynccnt 0" ::: "memory");
    __syncthreads();

    for (int o = tid; o < TT * HT * Wsz; o += 256) {
        const int tl = o >> 10;          // / (HT*Wsz)
        const int hl = (o >> 6) & (HT - 1);
        const int wl = o & (Wsz - 1);
        float sum = 0.f;
#pragma unroll
        for (int dt = 0; dt < KT; ++dt) {
            const int tb = (tl + 4 - dt) * HP;
#pragma unroll
            for (int dh = 0; dh < KS; ++dh) {
                const int hb = (tb + (hl + 6 - dh)) * WP + (wl + 6);
                const float* kr = &kS[(dt * KS + dh) * KS];
#pragma unroll
                for (int dw = 0; dw < KS; ++dw)
                    sum = fmaf(kr[dw], tileS[hb - dw], sum);
            }
        }
        dst[((((size_t)b * Tsz + (t0 + tl)) * Hsz + (h0 + hl)) * Wsz + wl) * Csz + c] = sum;
    }
}

// =====================================================================
// minGRU log-space scan over T, one thread per (b,h,w,c) chain.
// =====================================================================
__device__ __forceinline__ float softplusf(float x) {
    return fmaxf(x, 0.f) + log1pf(__expf(-fabsf(x)));
}
__device__ __forceinline__ float logaddexpf(float a, float b) {
    return fmaxf(a, b) + log1pf(__expf(-fabsf(a - b)));
}

__global__ __launch_bounds__(256) void scan_kernel(
    const float* __restrict__ gs, const float* __restrict__ hs,
    float* __restrict__ hout)
{
    const size_t id  = (size_t)blockIdx.x * 256 + threadIdx.x; // 524288 chains
    const int    c   = id & 63;
    const size_t r1  = id >> 6;
    const int    hwi = (int)(r1 & 4095);
    const int    b   = (int)(r1 >> 12);
    const size_t tstride = (size_t)Hsz * Wsz * Csz;
    const size_t base = (((size_t)b * Tsz) * Hsz * Wsz + hwi) * Csz + c;

    float lh = 0.f;
    for (int t = 0; t < Tsz; ++t) {
        const size_t off = base + (size_t)t * tstride;
        if (t + 1 < Tsz) {                       // hide the 1MB-stride walk
            __builtin_prefetch(gs + off + tstride, 0, 0);
            __builtin_prefetch(hs + off + tstride, 0, 0);
        }
        const float g = gs[off];
        const float v = hs[off];
        const float lf = -softplusf(g);                          // log(1-z)
        const float lv = -softplusf(-g) + __logf(v * v + 1e-6f); // log z + log h~
        lh = (t == 0) ? lv : logaddexpf(lf + lh, lv);
        hout[off] = __expf(lh);
    }
}

// =====================================================================
// Output projection: out = h @ out_w + out_b   (same WMMA scheme)
// =====================================================================
__global__ __launch_bounds__(256) void outproj_kernel(
    const float* __restrict__ hbuf,
    const float* __restrict__ ow, const float* __restrict__ ob,
    float* __restrict__ out)
{
    const int lane  = threadIdx.x & 31;
    const int wave  = threadIdx.x >> 5;
    const int gwv   = blockIdx.x * 8 + wave;
    const int tile0 = gwv * 8;
    const int ncol  = lane & 15;
    const int hi    = lane >> 4;

    v16h Bw[2][4];
    for (int ks = 0; ks < 2; ++ks) {
        const int klo = hi * 16 + ks * 32;
        for (int nt = 0; nt < 4; ++nt) {
            const int n = ncol + nt * 16;
            v16h bb;
#pragma unroll
            for (int j = 0; j < 16; ++j)
                bb[j] = (_Float16)ow[(size_t)(klo + j) * Csz + n];
            Bw[ks][nt] = bb;
        }
    }
    float bias[4];
#pragma unroll
    for (int nt = 0; nt < 4; ++nt) bias[nt] = ob[ncol + nt * 16];

    for (int r = 0; r < 8; ++r) {
        const int rowtile = tile0 + r;
        const size_t row  = (size_t)rowtile * 16 + ncol;
        const float* hr   = hbuf + row * Csz;

        v8f acc[4];
#pragma unroll
        for (int nt = 0; nt < 4; ++nt)
#pragma unroll
            for (int i = 0; i < 8; ++i) acc[nt][i] = 0.f;

#pragma unroll
        for (int ks = 0; ks < 2; ++ks) {
            const int kb = hi * 16 + ks * 32;
            v16h a;
#pragma unroll
            for (int j = 0; j < 16; ++j)
                a[j] = (_Float16)hr[kb + j];
#pragma unroll
            for (int nt = 0; nt < 4; ++nt)
                acc[nt] = __builtin_amdgcn_wmma_f32_16x16x32_f16(
                    false, a, false, Bw[ks][nt], (short)0, acc[nt], false, false);
        }

        const size_t mbase = (size_t)rowtile * 16 + hi * 8;
#pragma unroll
        for (int nt = 0; nt < 4; ++nt) {
            const int n = ncol + nt * 16;
#pragma unroll
            for (int v = 0; v < 8; ++v)
                out[(mbase + v) * Csz + n] = acc[nt][v] + bias[nt];
        }
    }
}

// =====================================================================
// host-side launcher
// inputs: x, gate_w, gate_b, hidden_w, hidden_b, gate_kernel, hidden_kernel,
//         out_w, out_b   (all float32)
// workspace layout: [gp | hp | gs | hs], 64 MiB each (256 MiB total);
// h reuses the gp slot after the conv stage.
// =====================================================================
extern "C" void kernel_launch(void* const* d_in, const int* in_sizes, int n_in,
                              void* d_out, int out_size, void* d_ws, size_t ws_size,
                              hipStream_t stream) {
    const float* x   = (const float*)d_in[0];
    const float* gw  = (const float*)d_in[1];
    const float* gb  = (const float*)d_in[2];
    const float* hw  = (const float*)d_in[3];
    const float* hb  = (const float*)d_in[4];
    const float* gk  = (const float*)d_in[5];
    const float* hk  = (const float*)d_in[6];
    const float* ow  = (const float*)d_in[7];
    const float* ob  = (const float*)d_in[8];
    float* out = (float*)d_out;

    float* gp = (float*)d_ws;
    float* hp = gp + NC;
    float* gs = hp + NC;
    float* hs = gs + NC;
    float* hbuf = gp;   // reuse

    // 2048 waves, 8 waves/block
    proj2_kernel<<<256, 256, 0, stream>>>(x, gw, gb, hw, hb, gp, hp);

    dim3 cgrid((Tsz / TT) * (Hsz / HT), Bsz * Csz);   // 32 x 128
    dwconv_kernel<<<cgrid, 256, 0, stream>>>(gp, gk, gs);
    dwconv_kernel<<<cgrid, 256, 0, stream>>>(hp, hk, hs);

    scan_kernel<<<(Bsz * Hsz * Wsz * Csz) / 256, 256, 0, stream>>>(gs, hs, hbuf);

    outproj_kernel<<<256, 256, 0, stream>>>(hbuf, ow, ob, out);
}
